// DetectionTargetLayer_50096498540733
// MI455X (gfx1250) — matrix operand order repaired and checked by reading, not compiled, not run on hardware
//
#include <hip/hip_runtime.h>

// ---------------------------------------------------------------------------
// DetectionTargetLayer for MI455X (gfx1250, wave32).
//   K1: per-proposal IoU max / crowd max / pos-neg flags        (B*N threads)
//   K2: exact lax.top_k via rank counting (value desc, idx asc) (B*N threads)
//   K3: one wave32 per output slot: argmax GT, deltas, and the
//       28x28 mask crop-resize expressed as Ay(28x56) x Tm(56x28)
//       on the matrix pipe: 8x v_wmma_f32_16x16x32_f16 per ROI.
// ---------------------------------------------------------------------------

namespace {
constexpr int kB = 2, kN = 2000, kG = 100, kH = 800, kW = 800;
constexpr int kT = 200, kP = 66, kNeg = 134, kMH = 28, kMW = 28;
}

typedef __attribute__((ext_vector_type(16))) _Float16 v16h;
typedef __attribute__((ext_vector_type(8)))  float    v8f;

__device__ __forceinline__ float mask_at(const int* m, int b, int y, int x, int g) {
  long long idx = (((long long)b * kH + y) * kW + x) * (long long)kG + g;
  return m[idx] != 0 ? 1.0f : 0.0f;
}

// ---------------- K1: per-proposal scores ----------------------------------
__global__ void dtl_scores(const float* __restrict__ props,
                           const int*   __restrict__ cls,
                           const float* __restrict__ boxes,
                           float* __restrict__ pos_score,
                           float* __restrict__ neg_score,
                           int* __restrict__ posflag,
                           int* __restrict__ negflag) {
  int t = blockIdx.x * blockDim.x + threadIdx.x;
  if (t >= kB * kN) return;
  int b = t / kN;
  const float* p = props + (long long)t * 4;
  float py1 = p[0], px1 = p[1], py2 = p[2], px2 = p[3];
  float pa = (py2 - py1) * (px2 - px1);
  float riou = -1e30f, crowd_max = -1e30f;
  for (int g = 0; g < kG; ++g) {
    const float* q = boxes + ((long long)b * kG + g) * 4;
    float yy1 = fmaxf(py1, q[0]);
    float xx1 = fmaxf(px1, q[1]);
    float yy2 = fminf(py2, q[2]);
    float xx2 = fminf(px2, q[3]);
    float inter = fmaxf(yy2 - yy1, 0.f) * fmaxf(xx2 - xx1, 0.f);
    float ga = (q[2] - q[0]) * (q[3] - q[1]);
    float o = inter / fmaxf(pa + ga - inter, 1e-12f);
    bool crowd = cls[b * kG + g] < 0;
    riou = fmaxf(riou, crowd ? -1.f : o);
    crowd_max = fmaxf(crowd_max, crowd ? o : -1.f);
  }
  bool pos = riou >= 0.5f;
  bool neg = (riou < 0.5f) && (crowd_max < 0.001f);
  pos_score[t] = pos ? riou : -1.f;
  neg_score[t] = neg ? riou : -1.f;
  posflag[t] = pos ? 1 : 0;
  negflag[t] = neg ? 1 : 0;
}

// ---------------- K2: stable top-K via rank counting -----------------------
__global__ void dtl_select(const float* __restrict__ score,
                           const int*   __restrict__ flag,
                           int* __restrict__ sel_idx,
                           int* __restrict__ sel_ok, int K) {
  int t = blockIdx.x * blockDim.x + threadIdx.x;
  if (t >= kB * kN) return;
  int b = t / kN, i = t % kN;
  const float* s = score + (long long)b * kN;
  float si = s[i];
  int rank = 0;
  for (int j = 0; j < kN; ++j) {
    float sj = s[j];
    rank += (int)((sj > si) || (sj == si && j < i));
  }
  if (rank < K) {
    sel_idx[b * K + rank] = i;
    sel_ok[b * K + rank]  = flag[t];
  }
}

// ---------------- K3: emit rois/class/deltas/masks (1 wave32 per slot) -----
__global__ void __launch_bounds__(32)
dtl_emit(const float* __restrict__ props,
         const int*   __restrict__ cls,
         const float* __restrict__ boxes,
         const int*   __restrict__ masks,
         const int* __restrict__ pos_idx, const int* __restrict__ pos_ok,
         const int* __restrict__ neg_idx, const int* __restrict__ neg_ok,
         float* __restrict__ out) {
  const int lane = threadIdx.x;
  const int bs   = blockIdx.x;
  const int b    = bs / kT;
  const int slot = bs % kT;

  float* rois_out = out;                                  // kB*kT*4
  float* cls_out  = out + kB * kT * 4;                    // kB*kT
  float* del_out  = out + kB * kT * 4 + kB * kT;          // kB*kT*4
  float* mask_out = out + kB * kT * 4 + kB * kT + kB * kT * 4;

  float* roi_row = rois_out + (size_t)(b * kT + slot) * 4;
  float* del_row = del_out  + (size_t)(b * kT + slot) * 4;
  float* m_row   = mask_out + (size_t)(b * kT + slot) * kMH * kMW;

  if (slot >= kP) {                       // negative slot (block-uniform path)
    if (lane == 0) {
      int ns  = slot - kP;
      int idx = neg_idx[b * kNeg + ns];
      int ok  = neg_ok[b * kNeg + ns];
      const float* p = props + (size_t)(b * kN + idx) * 4;
      for (int c = 0; c < 4; ++c) {
        roi_row[c] = ok ? p[c] : 0.f;
        del_row[c] = 0.f;
      }
      cls_out[b * kT + slot] = 0.f;
    }
    for (int k = lane; k < kMH * kMW; k += 32) m_row[k] = 0.f;
    return;
  }

  __shared__ float s_box[4];
  __shared__ float s_wy[kMH], s_wx[kMW];
  __shared__ int   s_y0[kMH], s_y1[kMH], s_x0[kMW], s_x1[kMW];
  __shared__ int   s_assign, s_ok;

  if (lane == 0) {
    int idx = pos_idx[b * kP + slot];
    int ok  = pos_ok[b * kP + slot];
    const float* p = props + (size_t)(b * kN + idx) * 4;
    float py1 = p[0], px1 = p[1], py2 = p[2], px2 = p[3];
    float pa = (py2 - py1) * (px2 - px1);
    float best = -1e30f; int bi = 0;
    for (int g = 0; g < kG; ++g) {            // argmax over crowd-masked IoU
      const float* q = boxes + (size_t)(b * kG + g) * 4;
      float yy1 = fmaxf(py1, q[0]);
      float xx1 = fmaxf(px1, q[1]);
      float yy2 = fminf(py2, q[2]);
      float xx2 = fminf(px2, q[3]);
      float inter = fmaxf(yy2 - yy1, 0.f) * fmaxf(xx2 - xx1, 0.f);
      float ga = (q[2] - q[0]) * (q[3] - q[1]);
      float o = inter / fmaxf(pa + ga - inter, 1e-12f);
      float ov = (cls[b * kG + g] < 0) ? -1.f : o;
      if (ov > best) { best = ov; bi = g; }   // first-max, matches jnp.argmax
    }
    const float* q = boxes + (size_t)(b * kG + bi) * 4;
    float h = py2 - py1, w = px2 - px1;
    float cy = py1 + 0.5f * h, cx = px1 + 0.5f * w;
    float gh = q[2] - q[0], gw = q[3] - q[1];
    float gcy = q[0] + 0.5f * gh, gcx = q[1] + 0.5f * gw;
    float d0 = ((gcy - cy) / h) / 0.1f;
    float d1 = ((gcx - cx) / w) / 0.1f;
    float d2 = logf(gh / h) / 0.2f;
    float d3 = logf(gw / w) / 0.2f;
    roi_row[0] = ok ? py1 : 0.f;  roi_row[1] = ok ? px1 : 0.f;
    roi_row[2] = ok ? py2 : 0.f;  roi_row[3] = ok ? px2 : 0.f;
    del_row[0] = ok ? d0 : 0.f;   del_row[1] = ok ? d1 : 0.f;
    del_row[2] = ok ? d2 : 0.f;   del_row[3] = ok ? d3 : 0.f;
    cls_out[b * kT + slot] = ok ? (float)cls[b * kG + bi] : 0.f;
    s_box[0] = py1; s_box[1] = px1; s_box[2] = py2; s_box[3] = px2;
    s_assign = bi; s_ok = ok;
  }
  __syncthreads();

  if (lane < kMH) {                         // interpolation tables (MH==MW)
    float by1 = s_box[0], bx1 = s_box[1], by2 = s_box[2], bx2 = s_box[3];
    float fy = (float)lane / (float)(kMH - 1);
    float ys = by1 * (kH - 1) + fy * (by2 - by1) * (kH - 1);
    float y0f = floorf(ys);
    s_wy[lane] = ys - y0f;
    int y0 = (int)y0f; y0 = y0 < 0 ? 0 : (y0 > kH - 1 ? kH - 1 : y0);
    int y1 = y0 + 1;   y1 = y1 > kH - 1 ? kH - 1 : y1;
    s_y0[lane] = y0; s_y1[lane] = y1;
    float fx = (float)lane / (float)(kMW - 1);
    float xs = bx1 * (kW - 1) + fx * (bx2 - bx1) * (kW - 1);
    float x0f = floorf(xs);
    s_wx[lane] = xs - x0f;
    int x0 = (int)x0f; x0 = x0 < 0 ? 0 : (x0 > kW - 1 ? kW - 1 : x0);
    int x1 = x0 + 1;   x1 = x1 > kW - 1 ? kW - 1 : x1;
    s_x0[lane] = x0; s_x1[lane] = x1;
  }
  __syncthreads();

  // out(28x28) = Ay(28x56) * Tm(56x28), padded to 32x64 * 64x32.
  // A operand (16-bit 16x32, ISA 7.12.2): lane m = lane&15; lanes>=16 carry
  // K+8 within each 16-K half; element e -> K = 16*(e>>3) + kb + 2*((e>>1)&3) + (e&1).
// B operand (32x16): N = lane&15; lanes>=16 carry K+8; element e -> K = 16*(e>>3) + kb + (e&7).
  const int assign = s_assign;
  const int half_i = lane & 15;
  const int a_kb   = (lane < 16) ? 0 : 8;
  const int b_kb   = (lane < 16) ? 0 : 8;

  v16h Aop[2][2];
  v16h Bop[2][2];
#pragma unroll
  for (int kc = 0; kc < 2; ++kc) {
#pragma unroll
    for (int mt = 0; mt < 2; ++mt) {
      v16h a;
#pragma unroll
      for (int e = 0; e < 16; ++e) {
        int M = mt * 16 + half_i;
        int K = kc * 32 + ((e >> 3) << 4) + a_kb + (((e >> 1) & 3) << 1) + (e & 1);
        float v = 0.f;
        if (M < kMH && (K >> 1) == M) v = (K & 1) ? s_wy[M] : 1.f - s_wy[M];
        a[e] = (_Float16)v;
      }
      Aop[mt][kc] = a;
    }
#pragma unroll
    for (int nt = 0; nt < 2; ++nt) {
      v16h bb;
      int Ncol = nt * 16 + half_i;
#pragma unroll
      for (int e = 0; e < 16; ++e) {
        int K = kc * 32 + ((e >> 3) << 4) + b_kb + (e & 7);           // row of Tm (0..55 valid)
        float v = 0.f;
        if (K < 2 * kMH && Ncol < kMW) {
          int i = K >> 1;
          int row = (K & 1) ? s_y1[i] : s_y0[i];
          float wx = s_wx[Ncol];
          float v0 = mask_at(masks, b, row, s_x0[Ncol], assign);
          float v1 = mask_at(masks, b, row, s_x1[Ncol], assign);
          v = v0 * (1.f - wx) + v1 * wx;      // x-interp (Tm entry)
        }
        bb[e] = (_Float16)v;
      }
      Bop[nt][kc] = bb;
    }
  }

  v8f acc[2][2] = {};
#pragma unroll
  for (int mt = 0; mt < 2; ++mt)
#pragma unroll
    for (int nt = 0; nt < 2; ++nt)
#pragma unroll
      for (int kc = 0; kc < 2; ++kc)
        acc[mt][nt] = __builtin_amdgcn_wmma_f32_16x16x32_f16(
            false, Aop[mt][kc], false, Bop[nt][kc],
            (short)0, acc[mt][nt], false, false);

  // C/D layout: VGPR r, lanes 0-15 -> M=r, lanes 16-31 -> M=r+8; N = lane&15.
  const int ok = s_ok;
#pragma unroll
  for (int mt = 0; mt < 2; ++mt)
#pragma unroll
    for (int nt = 0; nt < 2; ++nt) {
      v8f c = acc[mt][nt];
#pragma unroll
      for (int r = 0; r < 8; ++r) {
        int M = mt * 16 + r + ((lane >= 16) ? 8 : 0);
        int Ncol = nt * 16 + (lane & 15);
        if (M < kMH && Ncol < kMW)
          m_row[M * kMW + Ncol] = ok ? rintf(c[r]) : 0.f;  // RNE = jnp.round
      }
    }
}

// ---------------------------------------------------------------------------
extern "C" void kernel_launch(void* const* d_in, const int* in_sizes, int n_in,
                              void* d_out, int out_size, void* d_ws, size_t ws_size,
                              hipStream_t stream) {
  (void)in_sizes; (void)n_in; (void)out_size; (void)ws_size;
  const float* props = (const float*)d_in[0];
  const int*   cls   = (const int*)  d_in[1];
  const float* boxes = (const float*)d_in[2];
  const int*   masks = (const int*)  d_in[3];
  float* out = (float*)d_out;

  char* ws = (char*)d_ws;
  float* pos_score = (float*)ws; ws += sizeof(float) * kB * kN;
  float* neg_score = (float*)ws; ws += sizeof(float) * kB * kN;
  int* posflag = (int*)ws;       ws += sizeof(int) * kB * kN;
  int* negflag = (int*)ws;       ws += sizeof(int) * kB * kN;
  int* pos_idx = (int*)ws;       ws += sizeof(int) * kB * kP;
  int* pos_ok  = (int*)ws;       ws += sizeof(int) * kB * kP;
  int* neg_idx = (int*)ws;       ws += sizeof(int) * kB * kNeg;
  int* neg_ok  = (int*)ws;       ws += sizeof(int) * kB * kNeg;

  dim3 blk(256);
  dim3 grd((kB * kN + 255) / 256);
  dtl_scores<<<grd, blk, 0, stream>>>(props, cls, boxes,
                                      pos_score, neg_score, posflag, negflag);
  dtl_select<<<grd, blk, 0, stream>>>(pos_score, posflag, pos_idx, pos_ok, kP);
  dtl_select<<<grd, blk, 0, stream>>>(neg_score, negflag, neg_idx, neg_ok, kNeg);
  dtl_emit<<<dim3(kB * kT), dim3(32), 0, stream>>>(props, cls, boxes, masks,
                                                   pos_idx, pos_ok, neg_idx, neg_ok,
                                                   out);
}
